// SimpleMoEBlock_25314537242699
// MI455X (gfx1250) — matrix operand order repaired and checked by reading, not compiled, not run on hardware
//
#include <hip/hip_runtime.h>

#define H     1024
#define INTER 1408
#define NE    8
#define TM    64     // M tile (slots) per block
#define TN    128    // N tile per block
#define KSTEP 32     // bf16 WMMA K
#define RS    40     // padded LDS row stride in elements (80B): conflict-free frag reads

typedef __attribute__((ext_vector_type(16))) __bf16 bf16x16;
typedef __attribute__((ext_vector_type(8)))  __bf16 bf16x8;
typedef __attribute__((ext_vector_type(8)))  float  f32x8;

static __device__ __forceinline__ f32x8 wmma_bf16(bf16x16 a, bf16x16 b, f32x8 c) {
    // D = A(16x32 bf16) * B(32x16 bf16) + C(16x16 f32)
    return __builtin_amdgcn_wmma_f32_16x16x32_bf16(false, a, false, b, (short)0, c, false, false);
}

// Async DMA: 16B global -> LDS, tracked by ASYNCcnt (CDNA5 GLOBAL_LOAD_ASYNC_TO_LDS_B128).
static __device__ __forceinline__ void async_g2l_b128(unsigned lds_off, unsigned long long gaddr) {
    asm volatile("global_load_async_to_lds_b128 %0, %1, off" :: "v"(lds_off), "v"(gaddr) : "memory");
}
static __device__ __forceinline__ void wait_async0() {
    asm volatile("s_wait_asynccnt 0x0" ::: "memory");
}
// allow the newest batch (5 resp. 3 outstanding loads) to remain in flight
static __device__ __forceinline__ void wait_async5() {
    asm volatile("s_wait_asynccnt 0x5" ::: "memory");
}
static __device__ __forceinline__ void wait_async3() {
    asm volatile("s_wait_asynccnt 0x3" ::: "memory");
}

// A fragment (16x32 bf16): lane<16 holds row (lane&15), K {0..7,16..23};
// lane>=16 same row, K {8..15,24..31}. Two 16B LDS loads.
static __device__ __forceinline__ bf16x16 lds_a_frag(const __bf16* row, int khalf) {
    bf16x8 lo = *(const bf16x8*)(row + khalf * 8);
    bf16x8 hi = *(const bf16x8*)(row + khalf * 8 + 16);
    bf16x16 a;
#pragma unroll
    for (int i = 0; i < 8; ++i) { a[i] = lo[i]; a[i + 8] = hi[i]; }
    return a;
}
// B fragment (32x16 bf16): lane = N column, K {0..7,16..23} for lane<16, {8..15,24..31} for lane>=16.
static __device__ __forceinline__ bf16x16 lds_b_frag(const __bf16* row, int khalf) {
    bf16x8 lo = *(const bf16x8*)(row + khalf * 8);
    bf16x8 hi = *(const bf16x8*)(row + khalf * 8 + 16);
    bf16x16 b;
#pragma unroll
    for (int i = 0; i < 8; ++i) { b[i] = lo[i]; b[i + 8] = hi[i]; }
    return b;
}

// ---------------- conversion kernels ----------------

__global__ void cvt_x_bf16(const float* __restrict__ x, __bf16* __restrict__ xb, long n8) {
    long i = (long)blockIdx.x * blockDim.x + threadIdx.x;
    if (i >= n8) return;
    long base = i * 8;
    float4 v0 = *(const float4*)(x + base);
    float4 v1 = *(const float4*)(x + base + 4);
    bf16x8 o;
    o[0] = (__bf16)v0.x; o[1] = (__bf16)v0.y; o[2] = (__bf16)v0.z; o[3] = (__bf16)v0.w;
    o[4] = (__bf16)v1.x; o[5] = (__bf16)v1.y; o[6] = (__bf16)v1.z; o[7] = (__bf16)v1.w;
    *(bf16x8*)(xb + base) = o;
}

// per-expert transpose: in fp32 [R,C] -> out bf16 [C,R]
__global__ void transpose_cvt(const float* __restrict__ in, __bf16* __restrict__ out, int R, int C) {
    __shared__ float tile[32][33];
    size_t eoff = (size_t)blockIdx.z * R * C;
    const float* src = in + eoff;
    __bf16* dst = out + eoff;
    int r0 = blockIdx.x * 32, c0 = blockIdx.y * 32;
    int tx = threadIdx.x, ty = threadIdx.y;           // 32 x 8
#pragma unroll
    for (int i = 0; i < 32; i += 8) {
        int r = r0 + ty + i, c = c0 + tx;
        tile[ty + i][tx] = (r < R && c < C) ? src[(size_t)r * C + c] : 0.f;
    }
    __syncthreads();
#pragma unroll
    for (int i = 0; i < 32; i += 8) {
        int c = c0 + ty + i, r = r0 + tx;
        if (c < C && r < R) dst[(size_t)c * R + r] = (__bf16)tile[tx][ty + i];
    }
}

// ---------------- routing ----------------

__global__ void init_meta(int* __restrict__ counts, int* __restrict__ cursors,
                          int* __restrict__ slot_token, int slots) {
    int i = blockIdx.x * blockDim.x + threadIdx.x;
    if (i < NE) { counts[i] = 0; cursors[i] = 0; }
    if (i < slots) slot_token[i] = 0;
}

__global__ void router_kernel(const float* __restrict__ x, const float* __restrict__ gw,
                              float* __restrict__ logits, int* __restrict__ tok_e,
                              float* __restrict__ tok_w, int* __restrict__ counts, int T) {
    int lane = threadIdx.x & 31;
    int wid  = threadIdx.x >> 5;
    int t = blockIdx.x * 8 + wid;
    if (t >= T) return;
    const float* xr = x + (size_t)t * H;
    float acc[NE];
#pragma unroll
    for (int e = 0; e < NE; ++e) acc[e] = 0.f;
    for (int h = lane; h < H; h += 32) {
        float xv = xr[h];
#pragma unroll
        for (int e = 0; e < NE; ++e) acc[e] += xv * gw[e * H + h];
    }
#pragma unroll
    for (int e = 0; e < NE; ++e) {
#pragma unroll
        for (int off = 16; off >= 1; off >>= 1) acc[e] += __shfl_xor(acc[e], off, 32);
    }
    if (lane == 0) {
        float m = acc[0];
#pragma unroll
        for (int e = 1; e < NE; ++e) m = fmaxf(m, acc[e]);
        float p[NE], s = 0.f;
#pragma unroll
        for (int e = 0; e < NE; ++e) { p[e] = expf(acc[e] - m); s += p[e]; }
        float inv = 1.f / s;
#pragma unroll
        for (int e = 0; e < NE; ++e) logits[(size_t)t * NE + e] = acc[e];
        int i0 = 0;
#pragma unroll
        for (int e = 1; e < NE; ++e) if (acc[e] > acc[i0]) i0 = e;
        int i1 = (i0 == 0) ? 1 : 0;
#pragma unroll
        for (int e = 0; e < NE; ++e) if (e != i0 && acc[e] > acc[i1]) i1 = e;
        tok_e[2 * t]     = i0;  tok_w[2 * t]     = p[i0] * inv;
        tok_e[2 * t + 1] = i1;  tok_w[2 * t + 1] = p[i1] * inv;
        atomicAdd(&counts[i0], 1);
        atomicAdd(&counts[i1], 1);
    }
}

__global__ void offsets_kernel(const int* __restrict__ counts, int* __restrict__ exoff,
                               int* __restrict__ tileoff) {
    if (threadIdx.x == 0 && blockIdx.x == 0) {
        exoff[0] = 0; tileoff[0] = 0;
        for (int e = 0; e < NE; ++e) {
            int padded = ((counts[e] + TM - 1) / TM) * TM;
            exoff[e + 1] = exoff[e] + padded;
            tileoff[e + 1] = exoff[e + 1] / TM;
        }
    }
}

__global__ void scatter_kernel(const int* __restrict__ tok_e, int* __restrict__ cursors,
                               const int* __restrict__ exoff, int* __restrict__ slot_token,
                               int* __restrict__ tok_slot, int T) {
    int t = blockIdx.x * blockDim.x + threadIdx.x;
    if (t >= T) return;
#pragma unroll
    for (int k = 0; k < 2; ++k) {
        int e = tok_e[2 * t + k];
        int pos = atomicAdd(&cursors[e], 1);
        int slot = exoff[e] + pos;
        slot_token[slot] = t;
        tok_slot[2 * t + k] = slot;
    }
}

// ---------------- GEMM 1: fused gate+up, SiLU epilogue -> h (bf16) ----------------
// Async-staged LDS tiles: X(64x32) + Wg(128x32) + Wu(128x32) per k-step, TRIPLE buffered:
// batch for buffer i+2 issued at iter i; s_wait_asynccnt<=5 keeps newest batch in flight;
// one workgroup barrier per k-step. Regions per buffer (elems): X@0, G@2560, U@7680.
__global__ __launch_bounds__(256) void gemm_gate_up(
    const __bf16* __restrict__ xb, const __bf16* __restrict__ wgb, const __bf16* __restrict__ wub,
    const int* __restrict__ tileoff, const int* __restrict__ slot_token,
    __bf16* __restrict__ hbuf) {
    __shared__ __align__(16) __bf16 SB[3][12800];
    int mtile = blockIdx.x;
    int e = 0;
    while (e < NE && mtile >= tileoff[e + 1]) e++;
    if (e >= NE) return;
    int slotbase = mtile * TM;
    int n0 = blockIdx.y * TN;
    int tid = threadIdx.x;
    int lane = tid & 31, wid = tid >> 5;
    int wm = wid & 1, wn = wid >> 1;            // 2 M-waves x 4 N-waves
    int mlane = lane & 15, khalf = lane >> 4;
    size_t wbase = (size_t)e * H * INTER;

    // per-thread staging plan: 5 x 16B chunks per k-step (256+512+512 chunks)
    const __bf16* gsrc[5];
    unsigned lrel[5];
#pragma unroll
    for (int j = 0; j < 5; ++j) {
        int id = tid + 256 * j;
        if (id < 256) {
            int row = id >> 2, sub = id & 3;
            gsrc[j] = xb + (size_t)slot_token[slotbase + row] * H + sub * 8;
            lrel[j] = (unsigned)((row * RS + sub * 8) * 2);
        } else if (id < 768) {
            int q = id - 256, row = q >> 2, sub = q & 3;
            gsrc[j] = wgb + wbase + (size_t)(n0 + row) * H + sub * 8;
            lrel[j] = (unsigned)((2560 + row * RS + sub * 8) * 2);
        } else {
            int q = id - 768, row = q >> 2, sub = q & 3;
            gsrc[j] = wub + wbase + (size_t)(n0 + row) * H + sub * 8;
            lrel[j] = (unsigned)((7680 + row * RS + sub * 8) * 2);
        }
    }
    auto issue = [&](int b, int kelem) {
        unsigned lb = (unsigned)(unsigned long long)(&SB[b][0]);
#pragma unroll
        for (int j = 0; j < 5; ++j)
            async_g2l_b128(lb + lrel[j], (unsigned long long)(gsrc[j] + kelem));
    };

    f32x8 cg[2][2], cu[2][2];
    f32x8 z = {};
#pragma unroll
    for (int i = 0; i < 2; ++i)
#pragma unroll
        for (int j = 0; j < 2; ++j) { cg[i][j] = z; cu[i][j] = z; }

    const int nk = H / KSTEP;
    issue(0, 0);
    issue(1, KSTEP);
    int cur = 0;
    for (int i = 0; i < nk; ++i) {
        if (i + 1 < nk) wait_async5(); else wait_async0();  // current buffer's batch complete
        __syncthreads();                                    // publish to all waves; retire readers
        if (i + 2 < nk) {
            int nxt = cur + 2; if (nxt >= 3) nxt -= 3;
            issue(nxt, (i + 2) * KSTEP);
        }
        const __bf16* xs = &SB[cur][0];
        const __bf16* gs = &SB[cur][2560];
        const __bf16* us = &SB[cur][7680];
        bf16x16 a0 = lds_a_frag(xs + (wm * 32 + mlane) * RS, khalf);
        bf16x16 a1 = lds_a_frag(xs + (wm * 32 + 16 + mlane) * RS, khalf);
        bf16x16 g0 = lds_b_frag(gs + (wn * 32 + mlane) * RS, khalf);
        bf16x16 g1 = lds_b_frag(gs + (wn * 32 + 16 + mlane) * RS, khalf);
        bf16x16 u0 = lds_b_frag(us + (wn * 32 + mlane) * RS, khalf);
        bf16x16 u1 = lds_b_frag(us + (wn * 32 + 16 + mlane) * RS, khalf);
        cg[0][0] = wmma_bf16(a0, g0, cg[0][0]);
        cg[0][1] = wmma_bf16(a0, g1, cg[0][1]);
        cg[1][0] = wmma_bf16(a1, g0, cg[1][0]);
        cg[1][1] = wmma_bf16(a1, g1, cg[1][1]);
        cu[0][0] = wmma_bf16(a0, u0, cu[0][0]);
        cu[0][1] = wmma_bf16(a0, u1, cu[0][1]);
        cu[1][0] = wmma_bf16(a1, u0, cu[1][0]);
        cu[1][1] = wmma_bf16(a1, u1, cu[1][1]);
        cur = cur + 1; if (cur >= 3) cur = 0;
    }
#pragma unroll
    for (int mi = 0; mi < 2; ++mi)
#pragma unroll
        for (int ni = 0; ni < 2; ++ni)
#pragma unroll
            for (int r = 0; r < 8; ++r) {
                float gv = cg[mi][ni][r], uv = cu[mi][ni][r];
                float hv = (gv / (1.f + expf(-gv))) * uv;  // silu(g)*u
                int row = slotbase + wm * 32 + mi * 16 + r + 8 * khalf;
                int col = n0 + wn * 32 + ni * 16 + mlane;
                hbuf[(size_t)row * INTER + col] = (__bf16)hv;
            }
}

// ---------------- GEMM 2: down projection -> Y (fp32) ----------------
// Async-staged LDS tiles: h(64x32) + Wd(128x32), triple buffered (h@0, W@2560).
__global__ __launch_bounds__(256) void gemm_down(
    const __bf16* __restrict__ hbuf, const __bf16* __restrict__ wdb,
    const int* __restrict__ tileoff, float* __restrict__ Y) {
    __shared__ __align__(16) __bf16 SB[3][7680];
    int mtile = blockIdx.x;
    int e = 0;
    while (e < NE && mtile >= tileoff[e + 1]) e++;
    if (e >= NE) return;
    int slotbase = mtile * TM;
    int n0 = blockIdx.y * TN;
    int tid = threadIdx.x;
    int lane = tid & 31, wid = tid >> 5;
    int wm = wid & 1, wn = wid >> 1;
    int mlane = lane & 15, khalf = lane >> 4;
    size_t wbase = (size_t)e * H * INTER;

    const __bf16* gsrc[3];
    unsigned lrel[3];
#pragma unroll
    for (int j = 0; j < 3; ++j) {
        int id = tid + 256 * j;
        if (id < 256) {
            int row = id >> 2, sub = id & 3;
            gsrc[j] = hbuf + (size_t)(slotbase + row) * INTER + sub * 8;
            lrel[j] = (unsigned)((row * RS + sub * 8) * 2);
        } else {
            int q = id - 256, row = q >> 2, sub = q & 3;
            gsrc[j] = wdb + wbase + (size_t)(n0 + row) * INTER + sub * 8;
            lrel[j] = (unsigned)((2560 + row * RS + sub * 8) * 2);
        }
    }
    auto issue = [&](int b, int kelem) {
        unsigned lb = (unsigned)(unsigned long long)(&SB[b][0]);
#pragma unroll
        for (int j = 0; j < 3; ++j)
            async_g2l_b128(lb + lrel[j], (unsigned long long)(gsrc[j] + kelem));
    };

    f32x8 c[2][2];
    f32x8 z = {};
#pragma unroll
    for (int i = 0; i < 2; ++i)
#pragma unroll
        for (int j = 0; j < 2; ++j) c[i][j] = z;

    const int nk = INTER / KSTEP;
    issue(0, 0);
    issue(1, KSTEP);
    int cur = 0;
    for (int i = 0; i < nk; ++i) {
        if (i + 1 < nk) wait_async3(); else wait_async0();
        __syncthreads();
        if (i + 2 < nk) {
            int nxt = cur + 2; if (nxt >= 3) nxt -= 3;
            issue(nxt, (i + 2) * KSTEP);
        }
        const __bf16* xs = &SB[cur][0];
        const __bf16* wsld = &SB[cur][2560];
        bf16x16 a0 = lds_a_frag(xs + (wm * 32 + mlane) * RS, khalf);
        bf16x16 a1 = lds_a_frag(xs + (wm * 32 + 16 + mlane) * RS, khalf);
        bf16x16 f0 = lds_b_frag(wsld + (wn * 32 + mlane) * RS, khalf);
        bf16x16 f1 = lds_b_frag(wsld + (wn * 32 + 16 + mlane) * RS, khalf);
        c[0][0] = wmma_bf16(a0, f0, c[0][0]);
        c[0][1] = wmma_bf16(a0, f1, c[0][1]);
        c[1][0] = wmma_bf16(a1, f0, c[1][0]);
        c[1][1] = wmma_bf16(a1, f1, c[1][1]);
        cur = cur + 1; if (cur >= 3) cur = 0;
    }
#pragma unroll
    for (int mi = 0; mi < 2; ++mi)
#pragma unroll
        for (int ni = 0; ni < 2; ++ni)
#pragma unroll
            for (int r = 0; r < 8; ++r) {
                int row = slotbase + wm * 32 + mi * 16 + r + 8 * khalf;
                int col = n0 + wn * 32 + ni * 16 + mlane;
                Y[(size_t)row * H + col] = c[mi][ni][r];
            }
}

// ---------------- combine: out[t] = w0*Y[slot0] + w1*Y[slot1] (deterministic) ----------------
__global__ void combine_kernel(const float* __restrict__ Y, const int* __restrict__ tok_slot,
                               const float* __restrict__ tok_w, float* __restrict__ out, int T) {
    const int HV = H / 4;
    long idx = (long)blockIdx.x * blockDim.x + threadIdx.x;
    if (idx >= (long)T * HV) return;
    int t = (int)(idx / HV), cidx = (int)(idx % HV);
    int s0 = tok_slot[2 * t], s1 = tok_slot[2 * t + 1];
    float w0 = tok_w[2 * t], w1 = tok_w[2 * t + 1];
    float4 y0 = ((const float4*)(Y + (size_t)s0 * H))[cidx];
    float4 y1 = ((const float4*)(Y + (size_t)s1 * H))[cidx];
    float4 o;
    o.x = w0 * y0.x + w1 * y1.x;
    o.y = w0 * y0.y + w1 * y1.y;
    o.z = w0 * y0.z + w1 * y1.z;
    o.w = w0 * y0.w + w1 * y1.w;
    ((float4*)(out + (size_t)t * H))[cidx] = o;
}

// ---------------- launch ----------------

extern "C" void kernel_launch(void* const* d_in, const int* in_sizes, int n_in,
                              void* d_out, int out_size, void* d_ws, size_t ws_size,
                              hipStream_t stream) {
    const float* x      = (const float*)d_in[0];
    const float* gate_w = (const float*)d_in[1];
    const float* w_gate = (const float*)d_in[2];
    const float* w_up   = (const float*)d_in[3];
    const float* w_down = (const float*)d_in[4];
    float* out = (float*)d_out;

    const int T = in_sizes[0] / H;             // tokens
    const int SLOTS = T * 2 + NE * TM;         // worst-case padded slots
    float* logits = out + (size_t)T * H;

    // ---- workspace carve ----
    size_t cur = 0;
    auto alloc = [&](size_t bytes) {
        size_t o = cur;
        cur = (cur + bytes + 255) & ~(size_t)255;
        return o;
    };
    char* ws = (char*)d_ws;
    __bf16* xb  = (__bf16*)(ws + alloc((size_t)T * H * 2));
    __bf16* wgb = (__bf16*)(ws + alloc((size_t)NE * H * INTER * 2));
    __bf16* wub = (__bf16*)(ws + alloc((size_t)NE * H * INTER * 2));
    __bf16* wdb = (__bf16*)(ws + alloc((size_t)NE * H * INTER * 2));
    __bf16* hbuf = (__bf16*)(ws + alloc((size_t)SLOTS * INTER * 2));
    float*  Ybuf = (float*)(ws + alloc((size_t)SLOTS * H * 4));
    int* meta = (int*)(ws + alloc((size_t)(64 + 6 * T + SLOTS) * 4));
    int* counts   = meta;
    int* cursors  = meta + 8;
    int* exoff    = meta + 16;   // 9 entries
    int* tileoff  = meta + 32;   // 9 entries
    int* tok_e    = meta + 64;
    int* tok_slot = tok_e + 2 * T;
    float* tok_w  = (float*)(tok_slot + 2 * T);
    int* slot_token = (int*)(tok_w + 2 * T);
    (void)ws_size; (void)n_in; (void)out_size;

    // 1) precision conversion (weights transposed so B-fragments are K-contiguous)
    {
        long n8 = (long)T * H / 8;
        cvt_x_bf16<<<dim3((unsigned)((n8 + 255) / 256)), 256, 0, stream>>>(x, xb, n8);
        dim3 blk(32, 8);
        transpose_cvt<<<dim3(H / 32, INTER / 32, NE), blk, 0, stream>>>(w_gate, wgb, H, INTER);
        transpose_cvt<<<dim3(H / 32, INTER / 32, NE), blk, 0, stream>>>(w_up,   wub, H, INTER);
        transpose_cvt<<<dim3(INTER / 32, H / 32, NE), blk, 0, stream>>>(w_down, wdb, INTER, H);
    }
    // 2) routing
    init_meta<<<(SLOTS + 255) / 256, 256, 0, stream>>>(counts, cursors, slot_token, SLOTS);
    router_kernel<<<(T + 7) / 8, 256, 0, stream>>>(x, gate_w, logits, tok_e, tok_w, counts, T);
    offsets_kernel<<<1, 32, 0, stream>>>(counts, exoff, tileoff);
    scatter_kernel<<<(T + 255) / 256, 256, 0, stream>>>(tok_e, cursors, exoff, slot_token, tok_slot, T);
    // 3) expert GEMMs (WMMA bf16, f32 accumulate, async triple-buffered LDS staging)
    gemm_gate_up<<<dim3(SLOTS / TM, INTER / TN), 256, 0, stream>>>(xb, wgb, wub, tileoff, slot_token, hbuf);
    gemm_down<<<dim3(SLOTS / TM, H / TN), 256, 0, stream>>>(hbuf, wdb, tileoff, Ybuf);
    // 4) weighted combine (deterministic gather by per-token slots)
    combine_kernel<<<(unsigned)(((long)T * (H / 4) + 255) / 256), 256, 0, stream>>>(Ybuf, tok_slot, tok_w, out, T);
}
